// NetGINE_QM_ordered_28432683499902
// MI455X (gfx1250) — matrix operand order repaired and checked by reading, not compiled, not run on hardware
//
#include <hip/hip_runtime.h>
#include <math.h>

#define N_NODES 50000
#define N_EDGES 400000
#define NSUBG   2000
#define NGRAPH  250
#define DIMF    256
#define EXTRAF  64
#define XFEAT   16
#define EXFEAT  8
#define EFEAT   4
#define NCONV   4
#define STEPS   6
#define BN_EPS  1e-5f

typedef __attribute__((ext_vector_type(16))) __bf16 v16bf;
typedef __attribute__((ext_vector_type(8)))  float  v8f;

__device__ __forceinline__ unsigned int f2bf(float f) {
  unsigned int u = __float_as_uint(f);
  u += 0x7fffu + ((u >> 16) & 1u);          // round-to-nearest-even
  return u >> 16;
}
__device__ __forceinline__ unsigned int pack2bf(float a, float b) {
  return f2bf(a) | (f2bf(b) << 16);
}
__device__ __forceinline__ unsigned long long pack4bf(float4 v) {
  return (unsigned long long)pack2bf(v.x, v.y) |
         ((unsigned long long)pack2bf(v.z, v.w) << 32);
}

// ---------------------------------------------------------------------------
// Tiled bf16 WMMA GEMM: C[M,N] = (opt accum C) + A[M,K] * W + (opt bias), opt relu
// W layout: WT ? [N,K] row-major (ldw=K)  :  [K,N] row-major (ldw=N)
// Block: 256 threads (8 waves), C tile 128x64, wave tile 32x32 (2x2 wmma 16x16),
// K stepped by 32.  Double-buffered LDS software pipeline: next chunk's global
// b128 loads issue before this chunk's WMMAs; f32->bf16 conversion + ds_store
// happens after the WMMAs so global latency overlaps matrix math.
// Fragments use contiguous k = 16*half + s, the same order for A and B.
// ---------------------------------------------------------------------------
template <bool WT, bool BIAS, bool ACCUM, bool RELU>
__global__ __launch_bounds__(256) void gemm_bf16(
    const float* __restrict__ A, const float* __restrict__ W,
    const float* __restrict__ bias, float* __restrict__ C,
    int M, int N, int K, int ldw)
{
  __shared__ unsigned short As[2][128][36];   // [buf][m][k], 72B row stride
  __shared__ unsigned short Bs[2][64][36];    // [buf][n][k] (N-major)

  const int tid  = threadIdx.x;
  const int lane = tid & 31;
  const int w    = tid >> 5;
  const int wm   = w & 3;          // 0..3 -> M sub-tile
  const int wn   = w >> 2;         // 0..1 -> N sub-tile
  const int m0   = blockIdx.y * 128;
  const int n0   = blockIdx.x * 64;
  const int half = lane >> 4;      // lane 16..31 -> second K/M half
  const int l16  = lane & 15;

  // ---- per-thread staging tasks (iteration-invariant) ----
  int aR[4], aQ[4]; const float* aP[4]; bool aV[4];
  #pragma unroll
  for (int t = 0; t < 4; ++t) {
    int idx = tid + t * 256;            // 0..1023 : 128 rows x 8 quads
    aR[t] = idx >> 3; aQ[t] = idx & 7;
    int gr = m0 + aR[t];
    aV[t] = gr < M;
    int gra = aV[t] ? gr : (M - 1);     // clamped, always legal
    aP[t] = A + (long long)gra * K + 4 * aQ[t];
  }
  int bN[2], bQ[2]; const float* bP[2]; bool bV[2];
  #pragma unroll
  for (int t = 0; t < 2; ++t) {
    int idx = tid + t * 256;            // 0..511 : 64 n x 8 quads
    bN[t] = idx >> 3; bQ[t] = idx & 7;
    int gn = n0 + bN[t];
    bV[t] = gn < N;
    int gna = bV[t] ? gn : (N - 1);
    bP[t] = WT ? (W + (long long)gna * ldw + 4 * bQ[t])
               : (W + (long long)(4 * bQ[t]) * ldw + gna);
  }

  float4 aReg[4], bReg[2];
  auto loadG = [&]() {
    #pragma unroll
    for (int t = 0; t < 4; ++t) { aReg[t] = *(const float4*)aP[t]; aP[t] += 32; }
    #pragma unroll
    for (int t = 0; t < 2; ++t) {
      if (WT) {
        bReg[t] = *(const float4*)bP[t];
        bP[t] += 32;
      } else {
        bReg[t].x = bP[t][0];
        bReg[t].y = bP[t][(long long)ldw];
        bReg[t].z = bP[t][2 * (long long)ldw];
        bReg[t].w = bP[t][3 * (long long)ldw];
        bP[t] += 32 * (long long)ldw;
      }
    }
  };
  auto storeL = [&](int buf) {
    #pragma unroll
    for (int t = 0; t < 4; ++t) {
      float4 v = aReg[t];
      if (!aV[t]) { v.x = 0.f; v.y = 0.f; v.z = 0.f; v.w = 0.f; }
      *(unsigned long long*)&As[buf][aR[t]][4 * aQ[t]] = pack4bf(v);
    }
    #pragma unroll
    for (int t = 0; t < 2; ++t) {
      float4 v = bReg[t];
      if (!bV[t]) { v.x = 0.f; v.y = 0.f; v.z = 0.f; v.w = 0.f; }
      *(unsigned long long*)&Bs[buf][bN[t]][4 * bQ[t]] = pack4bf(v);
    }
  };

  v8f acc[2][2];
  acc[0][0] = (v8f){}; acc[0][1] = (v8f){};
  acc[1][0] = (v8f){}; acc[1][1] = (v8f){};

  const int nCh = K >> 5;
  loadG();
  storeL(0);
  for (int ic = 0; ic < nCh; ++ic) {
    __syncthreads();
    const bool more = (ic + 1) < nCh;
    if (more) loadG();                       // prefetch next chunk (global)

    const int buf = ic & 1;
    union { v16bf v; unsigned int u[8]; } af[2], bfrag[2];
    #pragma unroll
    for (int mi = 0; mi < 2; ++mi) {
      int row = wm * 32 + mi * 16 + l16;             // A: M = lane&15
      #pragma unroll
      for (int p = 0; p < 8; ++p) {
        int kb = half * 16 + 2 * p;                       // K layout, A 16x32
        af[mi].u[p] = *(const unsigned int*)&As[buf][row][kb];
      }
    }
    #pragma unroll
    for (int ni = 0; ni < 2; ++ni) {
      int col = wn * 32 + ni * 16 + l16;             // B: N = lane&15
      #pragma unroll
      for (int p = 0; p < 8; ++p) {
        int kb = half * 16 + 2 * p;                  // K layout, B 32x16
        bfrag[ni].u[p] = *(const unsigned int*)&Bs[buf][col][kb];
      }
    }
    #pragma unroll
    for (int mi = 0; mi < 2; ++mi)
      #pragma unroll
      for (int ni = 0; ni < 2; ++ni)
        acc[mi][ni] = __builtin_amdgcn_wmma_f32_16x16x32_bf16(
            false, af[mi].v, false, bfrag[ni].v, (short)0, acc[mi][ni],
            false, false);

    if (more) storeL(buf ^ 1);               // convert+store after the WMMAs
  }

  // epilogue: C/D layout -> M = r + 8*half, N = lane&15
  #pragma unroll
  for (int mi = 0; mi < 2; ++mi) {
    #pragma unroll
    for (int ni = 0; ni < 2; ++ni) {
      int colg = n0 + wn * 32 + ni * 16 + l16;
      if (colg >= N) continue;
      #pragma unroll
      for (int r = 0; r < 8; ++r) {
        int rowg = m0 + wm * 32 + mi * 16 + r + half * 8;
        if (rowg < M) {
          float v = acc[mi][ni][r];
          if (BIAS)  v += bias[colg];
          if (ACCUM) v += C[(long long)rowg * N + colg];
          if (RELU)  v = fmaxf(v, 0.0f);
          C[(long long)rowg * N + colg] = v;
        }
      }
    }
  }
}

// ---------------------------------------------------------------------------
// Elementwise / reduction helper kernels
// ---------------------------------------------------------------------------
__global__ void fill_kernel(float* p, float v, long long n) {
  long long i = (long long)blockIdx.x * 256 + threadIdx.x;
  if (i < n) p[i] = v;
}

__global__ void encoder_kernel(const float* __restrict__ xf,
                               const float* __restrict__ ex,
                               const float* __restrict__ Wx, const float* __restrict__ bx,
                               const float* __restrict__ We, const float* __restrict__ be,
                               float* __restrict__ X) {
  long long i = (long long)blockIdx.x * 256 + threadIdx.x;
  long long total = (long long)N_NODES * 320;
  if (i >= total) return;
  int n = (int)(i / 320), c = (int)(i % 320);
  float acc;
  if (c < DIMF) {
    acc = bx[c];
    const float* row = xf + (long long)n * XFEAT;
    #pragma unroll
    for (int k = 0; k < XFEAT; ++k) acc += row[k] * Wx[k * DIMF + c];
  } else {
    int c2 = c - DIMF;
    acc = be[c2];
    const float* row = ex + (long long)n * EXFEAT;
    #pragma unroll
    for (int k = 0; k < EXFEAT; ++k) acc += row[k] * We[k * EXTRAF + c2];
  }
  X[i] = fmaxf(acc, 0.0f);
}

__global__ void bond1_kernel(const float* __restrict__ ea,
                             const float* __restrict__ W, const float* __restrict__ b,
                             float* __restrict__ out, int d1) {
  long long i = (long long)blockIdx.x * 256 + threadIdx.x;
  long long total = (long long)N_EDGES * d1;
  if (i >= total) return;
  int e = (int)(i / d1), c = (int)(i % d1);
  const float* row = ea + (long long)e * EFEAT;
  float acc = b[c];
  #pragma unroll
  for (int k = 0; k < EFEAT; ++k) acc += row[k] * W[k * d1 + c];
  out[i] = acc;
}

__global__ void colstats_kernel(const float* __restrict__ X, int M, int C,
                                float* __restrict__ stats) {
  int r0 = blockIdx.x * 512;
  int r1 = r0 + 512; if (r1 > M) r1 = M;
  for (int c = threadIdx.x; c < C; c += 256) {
    float s = 0.0f, q = 0.0f;
    for (int r = r0; r < r1; ++r) {
      float v = X[(long long)r * C + c];
      s += v; q += v * v;
    }
    atomicAdd(&stats[c], s);
    atomicAdd(&stats[C + c], q);
  }
}

__global__ void bn_prep_kernel(float* __restrict__ stats,
                               const float* __restrict__ g, const float* __restrict__ B,
                               float invM, int C) {
  int c = blockIdx.x * 256 + threadIdx.x;
  if (c >= C) return;
  float mean = stats[c] * invM;
  float var  = stats[C + c] * invM - mean * mean;   // biased variance
  float sc = g[c] * rsqrtf(var + BN_EPS);
  stats[c] = sc;
  stats[C + c] = B[c] - mean * sc;
}

// vectorized: total4 = M*C/4 float4 elements
__global__ void bn_relu_kernel(float* __restrict__ X,
                               const float* __restrict__ stats,
                               long long total4, int C) {
  long long i = (long long)blockIdx.x * 256 + threadIdx.x;
  if (i >= total4) return;
  int q = (int)(i % (C / 4));
  float4 v = ((float4*)X)[i];
  float4 sc = *(const float4*)(stats + 4 * q);
  float4 sh = *(const float4*)(stats + C + 4 * q);
  v.x = fmaxf(v.x * sc.x + sh.x, 0.0f);
  v.y = fmaxf(v.y * sc.y + sh.y, 0.0f);
  v.z = fmaxf(v.z * sc.z + sh.z, 0.0f);
  v.w = fmaxf(v.w * sc.w + sh.w, 0.0f);
  ((float4*)X)[i] = v;
}

__global__ void agg_init_kernel(float* __restrict__ H, const float* __restrict__ X,
                                const float* __restrict__ eps, long long total4) {
  long long i = (long long)blockIdx.x * 256 + threadIdx.x;
  if (i >= total4) return;
  float s = 1.0f + eps[0];
  float4 v = ((const float4*)X)[i];
  v.x *= s; v.y *= s; v.z *= s; v.w *= s;
  ((float4*)H)[i] = v;
}

__global__ void scatter_kernel(float* __restrict__ H, const float* __restrict__ X,
                               const float* __restrict__ E2,
                               const int* __restrict__ src, const int* __restrict__ dst,
                               int d1) {
  long long i = (long long)blockIdx.x * 256 + threadIdx.x;
  int nq = d1 >> 2;
  long long total = (long long)N_EDGES * nq;
  if (i >= total) return;
  int e = (int)(i / nq), q = (int)(i % nq);
  float4 xe = *(const float4*)(X + (long long)src[e] * d1 + 4 * q);
  float4 ee = *(const float4*)(E2 + (long long)e * d1 + 4 * q);
  float* hp = H + (long long)dst[e] * d1 + 4 * q;
  atomicAdd(hp + 0, fmaxf(xe.x + ee.x, 0.0f));
  atomicAdd(hp + 1, fmaxf(xe.y + ee.y, 0.0f));
  atomicAdd(hp + 2, fmaxf(xe.z + ee.z, 0.0f));
  atomicAdd(hp + 3, fmaxf(xe.w + ee.w, 0.0f));
}

__global__ void mask_kernel(float* __restrict__ X, const float* __restrict__ m,
                            long long total4) {
  long long i = (long long)blockIdx.x * 256 + threadIdx.x;
  if (i >= total4) return;
  float s = m[i / (DIMF / 4)];
  float4 v = ((const float4*)X)[i];
  v.x *= s; v.y *= s; v.z *= s; v.w *= s;
  ((float4*)X)[i] = v;
}

__global__ void bias_sum_kernel(float* __restrict__ o, const float* __restrict__ a,
                                const float* __restrict__ b, int n) {
  int i = blockIdx.x * 256 + threadIdx.x;
  if (i < n) o[i] = a[i] + b[i];
}

__global__ void lstm_kernel(const float* __restrict__ G, float* __restrict__ H,
                            float* __restrict__ Cst) {
  int i = blockIdx.x * 256 + threadIdx.x;
  if (i >= NSUBG * DIMF) return;
  int s = i / DIMF, c = i % DIMF;
  const float* g = G + (long long)s * 4 * DIMF;
  float ig = 1.0f / (1.0f + __expf(-g[c]));
  float fg = 1.0f / (1.0f + __expf(-g[DIMF + c]));
  float gg = tanhf(g[2 * DIMF + c]);
  float og = 1.0f / (1.0f + __expf(-g[3 * DIMF + c]));
  float cn = fg * Cst[i] + ig * gg;
  Cst[i] = cn;
  H[i] = og * tanhf(cn);
}

__global__ void attn_kernel(const float* __restrict__ X, const float* __restrict__ H,
                            const int* __restrict__ batch, float* __restrict__ En) {
  int n = blockIdx.x * 256 + threadIdx.x;
  if (n >= N_NODES) return;
  const float4* xr = (const float4*)(X + (long long)n * DIMF);
  const float4* q  = (const float4*)(H + (long long)batch[n] * DIMF);
  float acc = 0.0f;
  #pragma unroll 8
  for (int k = 0; k < DIMF / 4; ++k) {
    float4 a = xr[k], b = q[k];
    acc += a.x * b.x + a.y * b.y + a.z * b.z + a.w * b.w;
  }
  En[n] = acc;
}

__global__ void segstart_kernel(const int* __restrict__ batch, int* __restrict__ segs) {
  int n = blockIdx.x * 256 + threadIdx.x;
  if (n >= N_NODES) return;
  int b = batch[n];
  int prev = (n == 0) ? -1 : batch[n - 1];
  for (int s = prev + 1; s <= b; ++s) segs[s] = n;
  if (n == N_NODES - 1)
    for (int s = b + 1; s <= NSUBG; ++s) segs[s] = N_NODES;
}

__global__ void segsoftmax_kernel(const float* __restrict__ En,
                                  const int* __restrict__ segs,
                                  float* __restrict__ An) {
  int s = blockIdx.x;
  int lo = segs[s], hi = segs[s + 1];
  int lane = threadIdx.x;
  float m = -INFINITY;
  for (int i = lo + lane; i < hi; i += 32) m = fmaxf(m, En[i]);
  for (int o = 16; o; o >>= 1) m = fmaxf(m, __shfl_xor(m, o, 32));
  if (!isfinite(m)) m = 0.0f;                       // matches reference emax fixup
  float sum = 0.0f;
  for (int i = lo + lane; i < hi; i += 32) {
    float ex = __expf(En[i] - m);
    An[i] = ex; sum += ex;
  }
  for (int o = 16; o; o >>= 1) sum += __shfl_xor(sum, o, 32);
  float inv = 1.0f / fmaxf(sum, 1e-16f);
  for (int i = lo + lane; i < hi; i += 32) An[i] *= inv;
}

__global__ void rqstar_kernel(const float* __restrict__ X, const float* __restrict__ An,
                              const int* __restrict__ segs, const float* __restrict__ H,
                              float* __restrict__ QS) {
  int s = blockIdx.x, c = threadIdx.x;
  int lo = segs[s], hi = segs[s + 1];
  float acc = 0.0f;
  for (int i = lo; i < hi; ++i) acc += An[i] * X[(long long)i * DIMF + c];
  QS[(long long)s * 2 * DIMF + c] = H[(long long)s * DIMF + c];
  QS[(long long)s * 2 * DIMF + DIMF + c] = acc;
}

__global__ void pool_kernel(const float* __restrict__ Y, const int* __restrict__ gidx,
                            float* __restrict__ GS) {
  int i = blockIdx.x * 256 + threadIdx.x;
  if (i >= NSUBG * DIMF) return;
  int s = i / DIMF, c = i % DIMF;
  atomicAdd(&GS[(long long)gidx[s] * DIMF + c], Y[i]);
}

__global__ void poolcnt_kernel(const int* __restrict__ gidx, float* __restrict__ GC) {
  int s = blockIdx.x * 256 + threadIdx.x;
  if (s >= NSUBG) return;
  atomicAdd(&GC[gidx[s]], 1.0f);
}

__global__ void final_kernel(const float* __restrict__ GS, const float* __restrict__ GC,
                             const float* __restrict__ W4, const float* __restrict__ b4,
                             float* __restrict__ out) {
  __shared__ float red[256];
  int g = blockIdx.x, c = threadIdx.x;
  float cnt = fmaxf(GC[g], 1.0f);
  red[c] = (GS[(long long)g * DIMF + c] / cnt) * W4[c];
  __syncthreads();
  for (int o = 128; o; o >>= 1) {
    if (c < o) red[c] += red[c + o];
    __syncthreads();
  }
  if (c == 0) out[g] = red[0] + b4[0];
}

// ---------------------------------------------------------------------------
// Host orchestration
// ---------------------------------------------------------------------------
static inline unsigned nb(long long n) { return (unsigned)((n + 255) / 256); }

extern "C" void kernel_launch(void* const* d_in, const int* in_sizes, int n_in,
                              void* d_out, int out_size, void* d_ws, size_t ws_size,
                              hipStream_t stream) {
  (void)in_sizes; (void)out_size; (void)ws_size;
  const float* x_feat = (const float*)d_in[0];
  const float* extra  = (const float*)d_in[1];
  const float* eattr  = (const float*)d_in[2];
  const float* nmask  = (const float*)d_in[3];

  // params: jax tree-flatten order (dict keys sorted; list in order).
  // per conv (17): bbn1.B, bbn1.g, bbn2.B, bbn2.g, bond1.W, bond1.b,
  //   bond2.W, bond2.b, eps, mbn1.B, mbn1.g, mbn2.B, mbn2.g,
  //   mlp1.W, mlp1.b, mlp2.W, mlp2.b
  // then enc_extra.{W,b}, enc_x.{W,b}, fc1.{W,b}, fc4.{W,b},
  //   lstm.{Whh,Wih,bhh,bih}
  const float* P[80];
  const int *edge_index, *batch, *gidx;
  if (n_in >= 87) {
    for (int i = 0; i < 80; ++i) P[i] = (const float*)d_in[4 + i];
    edge_index = (const int*)d_in[84];
    batch      = (const int*)d_in[85];
    gidx       = (const int*)d_in[86];
  } else {
    // fallback: params concatenated flat in d_in[4], same leaf order
    const float* base = (const float*)d_in[4];
    size_t off = 0; int pi = 0;
    for (int l = 0; l < NCONV; ++l) {
      size_t d1 = (l == 0) ? (DIMF + EXTRAF) : DIMF;
      size_t sz[17] = { d1, d1, d1, d1, 4 * d1, d1, d1 * d1, d1, 1,
                        DIMF, DIMF, DIMF, DIMF, d1 * DIMF, DIMF,
                        (size_t)DIMF * DIMF, DIMF };
      for (int j = 0; j < 17; ++j) { P[pi++] = base + off; off += sz[j]; }
    }
    size_t rest[12] = { (size_t)EXFEAT * EXTRAF, EXTRAF,
                        (size_t)XFEAT * DIMF, DIMF,
                        (size_t)2 * DIMF * DIMF, DIMF,
                        DIMF, 1,
                        (size_t)4 * DIMF * DIMF, (size_t)4 * DIMF * 2 * DIMF,
                        (size_t)4 * DIMF, (size_t)4 * DIMF };
    for (int j = 0; j < 12; ++j) { P[pi++] = base + off; off += rest[j]; }
    edge_index = (const int*)d_in[5];
    batch      = (const int*)d_in[6];
    gidx       = (const int*)d_in[7];
  }
  const int* src = edge_index;
  const int* dst = edge_index + N_EDGES;

  // workspace carve-out (16B aligned up to GC; GC's 254 floats leave BSUM/SEGS only 8B aligned)
  float* ws = (float*)d_ws;
  size_t cur = 0;
  auto carve = [&](size_t n) { float* p = ws + cur; cur += n; return p; };
  float* X    = carve((size_t)N_NODES * 320);
  float* H    = carve((size_t)N_NODES * 320);
  float* T    = carve((size_t)N_NODES * DIMF);
  float* EA   = carve((size_t)N_EDGES * 320);
  float* EB   = carve((size_t)N_EDGES * 320);
  float* ST   = carve(2048);
  float* QS   = carve((size_t)NSUBG * 2 * DIMF);
  float* HB   = carve((size_t)NSUBG * DIMF);
  float* CB   = carve((size_t)NSUBG * DIMF);
  float* GT   = carve((size_t)NSUBG * 4 * DIMF);
  float* EN   = carve(N_NODES);
  float* AN   = carve(N_NODES);
  float* Y    = carve((size_t)NSUBG * DIMF);
  float* GS   = carve((size_t)NGRAPH * DIMF);
  float* GC   = carve(NGRAPH + 4);
  float* BSUM = carve(4 * DIMF);
  int*   SEGS = (int*)carve(NSUBG + 4);

  // 1. node encoders -> X [N,320]
  encoder_kernel<<<nb((long long)N_NODES * 320), 256, 0, stream>>>(
      x_feat, extra, P[70], P[71], P[68], P[69], X);

  // 2. GIN conv layers
  for (int l = 0; l < NCONV; ++l) {
    const int d1 = (l == 0) ? (DIMF + EXTRAF) : DIMF;
    const float* const* p = P + l * 17;
    long long Etot = (long long)N_EDGES * d1;
    long long Ntot = (long long)N_NODES * d1;

    // bond encoder: lin(4->d1) -> BN -> relu
    bond1_kernel<<<nb(Etot), 256, 0, stream>>>(eattr, p[4], p[5], EA, d1);
    fill_kernel<<<nb(2 * d1), 256, 0, stream>>>(ST, 0.0f, 2 * d1);
    colstats_kernel<<<(N_EDGES + 511) / 512, 256, 0, stream>>>(EA, N_EDGES, d1, ST);
    bn_prep_kernel<<<nb(d1), 256, 0, stream>>>(ST, p[1], p[0], 1.0f / N_EDGES, d1);
    bn_relu_kernel<<<nb(Etot / 4), 256, 0, stream>>>(EA, ST, Etot / 4, d1);

    // lin(d1->d1) [WMMA] -> BN -> relu
    dim3 gE((d1 + 63) / 64, (N_EDGES + 127) / 128);
    gemm_bf16<false, true, false, false><<<gE, 256, 0, stream>>>(
        EA, p[6], p[7], EB, N_EDGES, d1, d1, d1);
    fill_kernel<<<nb(2 * d1), 256, 0, stream>>>(ST, 0.0f, 2 * d1);
    colstats_kernel<<<(N_EDGES + 511) / 512, 256, 0, stream>>>(EB, N_EDGES, d1, ST);
    bn_prep_kernel<<<nb(d1), 256, 0, stream>>>(ST, p[3], p[2], 1.0f / N_EDGES, d1);
    bn_relu_kernel<<<nb(Etot / 4), 256, 0, stream>>>(EB, ST, Etot / 4, d1);

    // h = (1+eps)*x + scatter_add(relu(x[src]+e), dst)
    agg_init_kernel<<<nb(Ntot / 4), 256, 0, stream>>>(H, X, p[8], Ntot / 4);
    scatter_kernel<<<nb(Etot / 4), 256, 0, stream>>>(H, X, EB, src, dst, d1);

    // mlp1 (d1->256) [WMMA] -> BN -> relu
    dim3 gN((DIMF + 63) / 64, (N_NODES + 127) / 128);
    gemm_bf16<false, true, false, false><<<gN, 256, 0, stream>>>(
        H, p[13], p[14], T, N_NODES, DIMF, d1, DIMF);
    long long NtotD = (long long)N_NODES * DIMF;
    fill_kernel<<<nb(2 * DIMF), 256, 0, stream>>>(ST, 0.0f, 2 * DIMF);
    colstats_kernel<<<(N_NODES + 511) / 512, 256, 0, stream>>>(T, N_NODES, DIMF, ST);
    bn_prep_kernel<<<nb(DIMF), 256, 0, stream>>>(ST, p[10], p[9], 1.0f / N_NODES, DIMF);
    bn_relu_kernel<<<nb(NtotD / 4), 256, 0, stream>>>(T, ST, NtotD / 4, DIMF);

    // mlp2 (256->256) [WMMA] -> BN -> relu (outer relu absorbed) -> X
    gemm_bf16<false, true, false, false><<<gN, 256, 0, stream>>>(
        T, p[15], p[16], X, N_NODES, DIMF, DIMF, DIMF);
    fill_kernel<<<nb(2 * DIMF), 256, 0, stream>>>(ST, 0.0f, 2 * DIMF);
    colstats_kernel<<<(N_NODES + 511) / 512, 256, 0, stream>>>(X, N_NODES, DIMF, ST);
    bn_prep_kernel<<<nb(DIMF), 256, 0, stream>>>(ST, p[12], p[11], 1.0f / N_NODES, DIMF);
    bn_relu_kernel<<<nb(NtotD / 4), 256, 0, stream>>>(X, ST, NtotD / 4, DIMF);
  }

  // 3. node mask
  mask_kernel<<<nb((long long)N_NODES * DIMF / 4), 256, 0, stream>>>(
      X, nmask, (long long)N_NODES * DIMF / 4);

  // 4. set2set
  fill_kernel<<<nb((long long)NSUBG * 2 * DIMF), 256, 0, stream>>>(QS, 0.0f, (long long)NSUBG * 2 * DIMF);
  fill_kernel<<<nb((long long)NSUBG * DIMF), 256, 0, stream>>>(HB, 0.0f, (long long)NSUBG * DIMF);
  fill_kernel<<<nb((long long)NSUBG * DIMF), 256, 0, stream>>>(CB, 0.0f, (long long)NSUBG * DIMF);
  bias_sum_kernel<<<nb(4 * DIMF), 256, 0, stream>>>(BSUM, P[79], P[78], 4 * DIMF);
  segstart_kernel<<<nb(N_NODES), 256, 0, stream>>>(batch, SEGS);

  for (int st = 0; st < STEPS; ++st) {
    dim3 gG((4 * DIMF + 63) / 64, (NSUBG + 127) / 128);
    // gates = q_star @ Wih^T + (bih + bhh)   [WMMA, W transposed access]
    gemm_bf16<true, true, false, false><<<gG, 256, 0, stream>>>(
        QS, P[77], BSUM, GT, NSUBG, 4 * DIMF, 2 * DIMF, 2 * DIMF);
    // gates += h @ Whh^T                     [WMMA accumulate]
    gemm_bf16<true, false, true, false><<<gG, 256, 0, stream>>>(
        HB, P[76], nullptr, GT, NSUBG, 4 * DIMF, DIMF, DIMF);
    lstm_kernel<<<nb(NSUBG * DIMF), 256, 0, stream>>>(GT, HB, CB);
    attn_kernel<<<nb(N_NODES), 256, 0, stream>>>(X, HB, batch, EN);
    segsoftmax_kernel<<<NSUBG, 32, 0, stream>>>(EN, SEGS, AN);
    rqstar_kernel<<<NSUBG, DIMF, 0, stream>>>(X, AN, SEGS, HB, QS);
  }

  // 5. fc1 + relu [WMMA]
  dim3 gF((DIMF + 63) / 64, (NSUBG + 127) / 128);
  gemm_bf16<false, true, false, true><<<gF, 256, 0, stream>>>(
      QS, P[72], P[73], Y, NSUBG, DIMF, 2 * DIMF, DIMF);

  // 6. global mean pool + fc4
  fill_kernel<<<nb((long long)NGRAPH * DIMF), 256, 0, stream>>>(GS, 0.0f, (long long)NGRAPH * DIMF);
  fill_kernel<<<nb(NGRAPH), 256, 0, stream>>>(GC, 0.0f, NGRAPH);
  pool_kernel<<<nb(NSUBG * DIMF), 256, 0, stream>>>(Y, gidx, GS);
  poolcnt_kernel<<<nb(NSUBG), 256, 0, stream>>>(gidx, GC);
  final_kernel<<<NGRAPH, DIMF, 0, stream>>>(GS, GC, P[74], P[75], (float*)d_out);
}
